// Attention_24670292148326
// MI455X (gfx1250) — matrix operand order repaired and checked by reading, not compiled, not run on hardware
//
#include <hip/hip_runtime.h>

#define N_TOK  8192
#define DIN    1024
#define DH     128
#define DOUT   1024
#define QKVN   (3 * DH)   // 384

typedef __attribute__((ext_vector_type(16))) __bf16 v16bf;
typedef __attribute__((ext_vector_type(8)))  __bf16 v8bf;
typedef __attribute__((ext_vector_type(8)))  float  v8f;
typedef __attribute__((ext_vector_type(4)))  float  f4;
typedef unsigned int u32x4 __attribute__((ext_vector_type(4)));
typedef int          i32x4 __attribute__((ext_vector_type(4)));
typedef int          i32x8 __attribute__((ext_vector_type(8)));

__device__ __forceinline__ __bf16 f2bf(float f) {
    unsigned u = __builtin_bit_cast(unsigned, f);
    unsigned r = u + 0x7FFFu + ((u >> 16) & 1u);       // round-to-nearest-even
    unsigned short h = (unsigned short)(r >> 16);
    return __builtin_bit_cast(__bf16, h);
}

// ---------------------------------------------------------------------------
// TDM: 2-D tile of 2-byte elements, global -> LDS.  Descriptor bit layout per
// CDNA5 ISA.  row_len/rows/stride in elements.
// This toolchain exposes the 6-arg builtin:
//   (uint32x4 g0, int32x8 g1, int32x4 g2, int32x4 g3, int32x8 extra, i32 cpol)
// ---------------------------------------------------------------------------
__device__ __forceinline__ void tdm_load_2d_bf16(unsigned lds_addr, const void* gptr,
                                                 unsigned row_len, unsigned rows,
                                                 unsigned stride)
{
    unsigned long long ga = (unsigned long long)(size_t)gptr;
    u32x4 g0;
    g0[0] = 1u;                                          // count=1, user D#
    g0[1] = lds_addr;                                    // lds_addr [63:32]
    g0[2] = (unsigned)(ga & 0xFFFFFFFFu);                // global_addr lo
    g0[3] = (unsigned)((ga >> 32) & 0x1FFFFFFu) | (2u << 30);  // ga hi | type=2
    i32x8 g1;
    g1[0] = (int)(1u << 16);          // wg_mask=0 | data_size=1 (2 bytes)
    g1[1] = (int)(row_len << 16);     // tensor_dim0[15:0] @ bits 63:48
    g1[2] = (int)(rows << 16);        // tensor_dim0 hi=0 | tensor_dim1 lo
    g1[3] = (int)(row_len << 16);     // tensor_dim1 hi=0 | tile_dim0
    g1[4] = (int)rows;                // tile_dim1 | tile_dim2=0
    g1[5] = (int)stride;              // tensor_dim0_stride lo32
    g1[6] = 0;                        // stride hi | dim1_stride lo
    g1[7] = 0;
    i32x4 z4;
    z4[0] = 0; z4[1] = 0; z4[2] = 0; z4[3] = 0;
    i32x8 z8;
    #pragma unroll
    for (int i = 0; i < 8; ++i) z8[i] = 0;
    __builtin_amdgcn_tensor_load_to_lds(g0, g1, z4, z4, z8, 0);
}

// ---------------------------------------------------------------------------
// Kernel 1: qkv = x @ w_qkv + b_qkv ; split into Q (pre-scaled), K (row-major)
// and V^T (transposed, [DH x N_TOK]) as bf16.  One wave -> one 16x16 tile.
// ---------------------------------------------------------------------------
__global__ __launch_bounds__(256) void qkv_proj_kernel(
    const float* __restrict__ x, const float* __restrict__ w,
    const float* __restrict__ bias,
    __bf16* __restrict__ Qb, __bf16* __restrict__ Kb, __bf16* __restrict__ Vt)
{
    const int lane = threadIdx.x & 31;
    const int wid  = threadIdx.x >> 5;
    const int gw   = blockIdx.x * 8 + wid;
    const int tiles_n = QKVN / 16;        // 24
    const int tm = gw / tiles_n;
    const int tn = gw % tiles_n;

    const int hlf  = lane >> 4;           // 0: lanes 0-15, 1: lanes 16-31
    const int l16  = lane & 15;
    const int aKlo = hlf * 8;             // unused; A frag also loads at bKlo
    const int bKlo = hlf * 16;            // A/B-frag K split: contiguous k = 16*hlf + s
    const int aRow = tm * 16 + l16;
    const int bCol = tn * 16 + l16;

    v8f c;
    #pragma unroll
    for (int i = 0; i < 8; ++i) c[i] = 0.0f;

    for (int kk = 0; kk < DIN; kk += 32) {
        const float* ap = x + (size_t)aRow * DIN + kk;
        f4 x0 = *(const f4*)(ap + bKlo);
        f4 x1 = *(const f4*)(ap + bKlo + 4);
        f4 x2 = *(const f4*)(ap + bKlo + 8);
        f4 x3 = *(const f4*)(ap + bKlo + 12);
        v16bf a, b;
        #pragma unroll
        for (int h = 0; h < 4; ++h) {
            a[h]      = f2bf(x0[h]);
            a[4 + h]  = f2bf(x1[h]);
            a[8 + h]  = f2bf(x2[h]);
            a[12 + h] = f2bf(x3[h]);
        }
        #pragma unroll
        for (int h = 0; h < 16; ++h) b[h] = f2bf(w[(size_t)(kk + bKlo + h) * QKVN + bCol]);
        c = __builtin_amdgcn_wmma_f32_16x16x32_bf16(false, a, false, b,
                                                    (short)0, c, false, false);
    }

    const float bv = bias[bCol];
    const float qscale = 0.08838834764831845f;   // 1/sqrt(128), baked into Q
    #pragma unroll
    for (int r = 0; r < 8; ++r) {
        int row = tm * 16 + r + hlf * 8;         // C/D layout: rows r / r+8
        float v = c[r] + bv;
        if (bCol < DH) {
            Qb[(size_t)row * DH + bCol] = f2bf(v * qscale);
        } else if (bCol < 2 * DH) {
            Kb[(size_t)row * DH + (bCol - DH)] = f2bf(v);
        } else {
            Vt[(size_t)(bCol - 2 * DH) * N_TOK + row] = f2bf(v);   // transposed
        }
    }
}

// ---------------------------------------------------------------------------
// Kernel 2: causal flash attention, block-cooperative.
// 8 waves / block, each wave owns 16 query rows; the block shares a TDM-staged
// KV chunk (32 keys) in LDS per iteration.  Online softmax; P transposed via
// per-wave LDS scratch.
// ---------------------------------------------------------------------------
__global__ __launch_bounds__(256) void flash_attn_kernel(
    const __bf16* __restrict__ Qb, const __bf16* __restrict__ Kb,
    const __bf16* __restrict__ Vt, __bf16* __restrict__ Ob)
{
    __shared__ __align__(32) __bf16 ldsK[32][DH];       // K chunk   (8 KB)
    __shared__ __align__(32) __bf16 ldsV[DH][32];       // V^T chunk (8 KB)
    __shared__ __align__(32) __bf16 plds[8][16][48];    // P transpose (12 KB)

    const int lane = threadIdx.x & 31;
    const int wid  = threadIdx.x >> 5;
    const int q0   = (blockIdx.x * 8 + wid) * 16;
    const int kvEnd = blockIdx.x * 128 + 128;           // block-uniform bound

    const int hlf  = lane >> 4;
    const int l16  = lane & 15;
    const int aKlo = hlf * 8;
    const int bKlo = hlf * 16;

    // Q fragments: 4 chunks of 32 over DH=128 (A layout: lane = row)
    v16bf qf[4];
    {
        const __bf16* qp = Qb + (size_t)(q0 + l16) * DH;
        #pragma unroll
        for (int t = 0; t < 4; ++t) {
            v8bf lo = *(const v8bf*)(qp + t * 32 + bKlo);
            v8bf hi = *(const v8bf*)(qp + t * 32 + bKlo + 8);
            #pragma unroll
            for (int h = 0; h < 8; ++h) { qf[t][h] = lo[h]; qf[t][8 + h] = hi[h]; }
        }
    }

    float mrow[8], lrow[8];
    v8f o[8];
    #pragma unroll
    for (int r = 0; r < 8; ++r) { mrow[r] = -3.0e38f; lrow[r] = 0.0f; }
    #pragma unroll
    for (int t = 0; t < 8; ++t) {
        #pragma unroll
        for (int r = 0; r < 8; ++r) o[t][r] = 0.0f;
    }

    for (int kv0 = 0; kv0 < kvEnd; kv0 += 32) {
        __syncthreads();                 // previous chunk fully consumed
        if (wid == 0) {
            // K tile: 32 rows x 128 cols, row stride DH elements
            tdm_load_2d_bf16((unsigned)(size_t)&ldsK[0][0],
                             (const void*)(Kb + (size_t)kv0 * DH),
                             DH, 32, DH);
            // V^T tile: 128 rows x 32 cols, row stride N_TOK elements
            tdm_load_2d_bf16((unsigned)(size_t)&ldsV[0][0],
                             (const void*)(Vt + kv0),
                             32, DH, N_TOK);
            __builtin_amdgcn_s_wait_tensorcnt(0);
        }
        __syncthreads();                 // chunk visible to all waves

        if (kv0 < q0 + 16) {             // wave-uniform causal guard
            // ---- S = Q K^T : two 16x16 sub-tiles
            v8f s[2];
            #pragma unroll
            for (int sub = 0; sub < 2; ++sub) {
                v8f acc;
                #pragma unroll
                for (int i = 0; i < 8; ++i) acc[i] = 0.0f;
                const __bf16* kp = &ldsK[sub * 16 + l16][0];   // lane = key col
                #pragma unroll
                for (int t = 0; t < 4; ++t) {
                    v16bf b = *(const v16bf*)(kp + t * 32 + bKlo);
                    acc = __builtin_amdgcn_wmma_f32_16x16x32_bf16(
                        false, qf[t], false, b, (short)0, acc, false, false);
                }
                s[sub] = acc;
            }

            // ---- causal mask (chunks touching the diagonal)
            if (kv0 + 31 > q0) {
                #pragma unroll
                for (int r = 0; r < 8; ++r) {
                    int i  = q0 + r + hlf * 8;
                    int j0 = kv0 + l16;
                    if (j0 > i)      s[0][r] = -1.0e10f;
                    if (j0 + 16 > i) s[1][r] = -1.0e10f;
                }
            }

            // ---- online softmax update
            #pragma unroll
            for (int r = 0; r < 8; ++r) {
                float cmax = fmaxf(s[0][r], s[1][r]);
                #pragma unroll
                for (int mk = 8; mk >= 1; mk >>= 1)
                    cmax = fmaxf(cmax, __shfl_xor(cmax, mk, 32));
                float newm  = fmaxf(mrow[r], cmax);
                float alpha = __expf(mrow[r] - newm);
                mrow[r] = newm;
                float p0 = __expf(s[0][r] - newm);
                float p1 = __expf(s[1][r] - newm);
                float ps = p0 + p1;
                #pragma unroll
                for (int mk = 8; mk >= 1; mk >>= 1)
                    ps += __shfl_xor(ps, mk, 32);
                lrow[r] = lrow[r] * alpha + ps;
                #pragma unroll
                for (int t = 0; t < 8; ++t) o[t][r] *= alpha;
                int rowl = r + hlf * 8;            // D layout: lane = column
                plds[wid][rowl][l16]      = f2bf(p0);
                plds[wid][rowl][16 + l16] = f2bf(p1);
            }
            asm volatile("s_wait_dscnt 0x0" ::: "memory");  // same-wave xpose

            // ---- reload P as a 16x32 A-fragment (lane = row)
            v16bf pf;
            {
                v8bf lo = *(const v8bf*)&plds[wid][l16][bKlo];
                v8bf hi = *(const v8bf*)&plds[wid][l16][bKlo + 8];
                #pragma unroll
                for (int h = 0; h < 8; ++h) { pf[h] = lo[h]; pf[8 + h] = hi[h]; }
            }

            // ---- O += P @ V  (8 output column tiles of 16)
            #pragma unroll
            for (int t = 0; t < 8; ++t) {
                v16bf b = *(const v16bf*)(&ldsV[t * 16 + l16][bKlo]);
                o[t] = __builtin_amdgcn_wmma_f32_16x16x32_bf16(
                    false, pf, false, b, (short)0, o[t], false, false);
            }
        }
    }

    // ---- finalize: O /= l, store bf16
    #pragma unroll
    for (int r = 0; r < 8; ++r) {
        float inv = 1.0f / lrow[r];
        int row = q0 + r + hlf * 8;
        #pragma unroll
        for (int t = 0; t < 8; ++t)
            Ob[(size_t)row * DH + t * 16 + l16] = f2bf(o[t][r] * inv);
    }
}

// ---------------------------------------------------------------------------
// Kernel 3: out = O @ w_out + b_out   ([8192x128] x [128x1024] -> f32)
// ---------------------------------------------------------------------------
__global__ __launch_bounds__(256) void out_proj_kernel(
    const __bf16* __restrict__ Ob, const float* __restrict__ w,
    const float* __restrict__ bias, float* __restrict__ out)
{
    const int lane = threadIdx.x & 31;
    const int wid  = threadIdx.x >> 5;
    const int gw   = blockIdx.x * 8 + wid;
    const int tiles_n = DOUT / 16;        // 64
    const int tm = gw / tiles_n;
    const int tn = gw % tiles_n;

    const int hlf  = lane >> 4;
    const int l16  = lane & 15;
    const int aKlo = hlf * 8;
    const int bKlo = hlf * 16;
    const int aRow = tm * 16 + l16;
    const int bCol = tn * 16 + l16;

    v8f c;
    #pragma unroll
    for (int i = 0; i < 8; ++i) c[i] = 0.0f;

    #pragma unroll
    for (int kk = 0; kk < DH; kk += 32) {
        const __bf16* ap = Ob + (size_t)aRow * DH + kk;
        v8bf lo = *(const v8bf*)(ap + bKlo);
        v8bf hi = *(const v8bf*)(ap + bKlo + 8);
        v16bf a, b;
        #pragma unroll
        for (int h = 0; h < 8; ++h) { a[h] = lo[h]; a[8 + h] = hi[h]; }
        #pragma unroll
        for (int h = 0; h < 16; ++h) b[h] = f2bf(w[(size_t)(kk + bKlo + h) * DOUT + bCol]);
        c = __builtin_amdgcn_wmma_f32_16x16x32_bf16(false, a, false, b,
                                                    (short)0, c, false, false);
    }

    const float bv = bias[bCol];
    #pragma unroll
    for (int r = 0; r < 8; ++r)
        out[(size_t)(tm * 16 + r + hlf * 8) * DOUT + bCol] = c[r] + bv;
}

// ---------------------------------------------------------------------------
extern "C" void kernel_launch(void* const* d_in, const int* in_sizes, int n_in,
                              void* d_out, int out_size, void* d_ws, size_t ws_size,
                              hipStream_t stream) {
    (void)in_sizes; (void)n_in; (void)out_size; (void)ws_size;

    const float* x      = (const float*)d_in[0];
    const float* w_qkv  = (const float*)d_in[1];
    const float* b_qkv  = (const float*)d_in[2];
    const float* w_out  = (const float*)d_in[3];
    const float* b_out  = (const float*)d_in[4];
    float* out          = (float*)d_out;

    // workspace: Q, K, V^T, O as bf16 [8192 x 128] each (8 MB total)
    __bf16* Qb = (__bf16*)d_ws;
    __bf16* Kb = Qb + (size_t)N_TOK * DH;
    __bf16* Vt = Kb + (size_t)N_TOK * DH;
    __bf16* Ob = Vt + (size_t)N_TOK * DH;

    qkv_proj_kernel<<<1536, 256, 0, stream>>>(x, w_qkv, b_qkv, Qb, Kb, Vt);
    flash_attn_kernel<<<64, 256, 0, stream>>>(Qb, Kb, Vt, Ob);
    out_proj_kernel<<<4096, 256, 0, stream>>>(Ob, w_out, b_out, out);
}
